// GNNReconstructor_73409581023296
// MI455X (gfx1250) — matrix operand order repaired and checked by reading, not compiled, not run on hardware
//
#include <hip/hip_runtime.h>
#include <hip/hip_bf16.h>

// ---------------- types matching CDNA5 WMMA builtin signatures ----------------
typedef __bf16 bf16_t;
typedef bf16_t v16bf __attribute__((ext_vector_type(16)));
typedef float  v8f   __attribute__((ext_vector_type(8)));
typedef float  v2f   __attribute__((ext_vector_type(2)));

#define M_N     20000
#define C_N     5000
#define D_K     128
#define NT_ROW  1250          // M/16
#define NT_COL  313           // ceil(5008/16): cols padded 5000 -> 5008
#define NCHUNK  8             // column chunks for partial softmax stats
#define TPC     40            // col tiles per chunk (8*40 = 320 >= 313)
#define NEG_HUGE (-3.402823466e38f)

// ---------------- ws layout (bytes) ----------------
#define OFF_XM   0u           // 20000*128*4  = 10,240,000
#define OFF_XC   10240000u    //  5000*128*4  =  2,560,000
#define OFF_AHI  12800000u    // 20000*128*2  =  5,120,000
#define OFF_ALO  17920000u
#define OFF_BHI  23040000u    //  5008*128*2  =  1,282,048
#define OFF_BLO  24322048u
#define OFF_SMAX 25604096u    // 20000*8*4    =    640,000
#define OFF_SSUM 26244096u
#define OFF_ROWM 26884096u    // 20000*4
#define OFF_ROWI 26964096u

// ---------------- kernels ----------------

__global__ void k_zero(float* __restrict__ p, int n) {
  int i = blockIdx.x * blockDim.x + threadIdx.x;
  if (i < n) p[i] = 0.f;
}

// x_m[row] += val * c_features[col]  (one wave per edge, 4 floats/lane)
__global__ void k_scatter_own(const float* __restrict__ c_feat,
                              const float* __restrict__ val,
                              const int* __restrict__ row,
                              const int* __restrict__ col,
                              float* __restrict__ xm, int ne) {
  int e = blockIdx.x * (blockDim.x >> 5) + (threadIdx.x >> 5);
  int lane = threadIdx.x & 31;
  if (e >= ne) return;
  int r = row[e], c = col[e];
  float v = val[e];
  float4 f = ((const float4*)(c_feat + (size_t)c * D_K))[lane];
  float* dst = xm + (size_t)r * D_K + lane * 4;
  atomicAdd(dst + 0, v * f.x);
  atomicAdd(dst + 1, v * f.y);
  atomicAdd(dst + 2, v * f.z);
  atomicAdd(dst + 3, v * f.w);
}

// Y[20000x128] += X @ W + bias      (exact f32 WMMA 16x16x4, 1 wave per 16x16 tile)
__global__ void k_lin_xm(const float* __restrict__ X, const float* __restrict__ W,
                         const float* __restrict__ bias, float* __restrict__ Y) {
  int wave = blockIdx.x * (blockDim.x >> 5) + (threadIdx.x >> 5); // 10000 waves
  int lane = threadIdx.x & 31;
  int rt = wave >> 3, ct = wave & 7;
  int half = lane >> 4, q16 = lane & 15;
  int m = rt * 16 + q16;
  int n = ct * 16 + q16;
  v8f acc = {};
#pragma unroll 4
  for (int kk = 0; kk < 32; ++kk) {
    int k0 = kk * 4 + half * 2;              // A 16x4 f32: lanes0-15 K0,K1; lanes16-31 K2,K3
    const float* xp = X + (size_t)m * D_K + k0;
    v2f a, b;
    a[0] = xp[0]; a[1] = xp[1];
    b[0] = W[(size_t)k0 * D_K + n];          // B 4x16 f32 (symmetric layout)
    b[1] = W[(size_t)(k0 + 1) * D_K + n];
    acc = __builtin_amdgcn_wmma_f32_16x16x4_f32(false, a, false, b,
                                                (short)0, acc, false, false);
  }
  float bv = bias[n];
#pragma unroll
  for (int r = 0; r < 8; ++r) {
    int rowi = rt * 16 + half * 8 + r;       // C layout: M = 8*(lane/16)+r, N = lane%16
    Y[(size_t)rowi * D_K + n] += acc[r] + bv;
  }
}

// x_c[5000x128] = X @ (W2+W3) + (b2+b3)   (313 row tiles, last tile partially masked)
__global__ void k_lin_xc(const float* __restrict__ X,
                         const float* __restrict__ W2, const float* __restrict__ W3,
                         const float* __restrict__ b2, const float* __restrict__ b3,
                         float* __restrict__ Y) {
  int wave = blockIdx.x * (blockDim.x >> 5) + (threadIdx.x >> 5); // 2504 waves exactly
  int lane = threadIdx.x & 31;
  int rt = wave >> 3, ct = wave & 7;
  int half = lane >> 4, q16 = lane & 15;
  int m = rt * 16 + q16; if (m > C_N - 1) m = C_N - 1;  // clamp loads, mask stores
  int n = ct * 16 + q16;
  v8f acc = {};
#pragma unroll 4
  for (int kk = 0; kk < 32; ++kk) {
    int k0 = kk * 4 + half * 2;
    const float* xp = X + (size_t)m * D_K + k0;
    v2f a, b;
    a[0] = xp[0]; a[1] = xp[1];
    b[0] = W2[(size_t)k0 * D_K + n] + W3[(size_t)k0 * D_K + n];
    b[1] = W2[(size_t)(k0 + 1) * D_K + n] + W3[(size_t)(k0 + 1) * D_K + n];
    acc = __builtin_amdgcn_wmma_f32_16x16x4_f32(false, a, false, b,
                                                (short)0, acc, false, false);
  }
  float bv = b2[n] + b3[n];
#pragma unroll
  for (int r = 0; r < 8; ++r) {
    int rowi = rt * 16 + half * 8 + r;
    if (rowi < C_N) Y[(size_t)rowi * D_K + n] = acc[r] + bv;
  }
}

// Pack x_m into bf16 hi/lo in the exact ISA 16-bit A-matrix 16x32 VGPR layout.
// A layout (§7.12.2): v<4: K=2v+8*(lane/16)+j ; v>=4: K=16+2(v-4)+8*(lane/16)+j
__global__ void k_pack_A(const float* __restrict__ Xm,
                         bf16_t* __restrict__ Ahi, bf16_t* __restrict__ Alo) {
  int tid = blockIdx.x * blockDim.x + threadIdx.x;
  if (tid >= NT_ROW * 32) return;
  int t = tid >> 5, lane = tid & 31;
  int m = t * 16 + (lane & 15);
  int half = lane >> 4;
#pragma unroll
  for (int qc = 0; qc < 4; ++qc) {
    size_t base = ((size_t)(t * 4 + qc) * 32 + lane) * 16;
#pragma unroll
    for (int e = 0; e < 16; ++e) {
      int v = e >> 1, j = e & 1;
      int k = qc * 32 + ((v < 4) ? 2 * v : 16 + 2 * (v - 4)) + half * 8 + j;
      float x = Xm[(size_t)m * D_K + k];
      bf16_t h = (bf16_t)x;
      bf16_t l = (bf16_t)(x - (float)h);
      Ahi[base + e] = h;
      Alo[base + e] = l;
    }
  }
}

// Pack x_c (as B = x_c^T) into bf16 hi/lo in the ISA 16-bit B-matrix 32x16 layout:
// lane%16 = N (= x_c row), v<4: K=2v+8*(lane/16)+j ; v>=4: K=16+2(v-4)+8*(lane/16)+j within each K=32 chunk.
__global__ void k_pack_B(const float* __restrict__ Xc,
                         bf16_t* __restrict__ Bhi, bf16_t* __restrict__ Blo) {
  int tid = blockIdx.x * blockDim.x + threadIdx.x;
  if (tid >= NT_COL * 32) return;
  int t = tid >> 5, lane = tid & 31;
  int n = t * 16 + (lane & 15);
  int half = lane >> 4;
#pragma unroll
  for (int qc = 0; qc < 4; ++qc) {
    size_t base = ((size_t)(t * 4 + qc) * 32 + lane) * 16;
#pragma unroll
    for (int e = 0; e < 16; ++e) {
      int v = e >> 1, j = e & 1;
      int k = qc * 32 + ((v < 4) ? 2 * v : 16 + 2 * (v - 4)) + half * 8 + j;
      float x = (n < C_N) ? Xc[(size_t)n * D_K + k] : 0.f;
      bf16_t h = (bf16_t)x;
      bf16_t l = (bf16_t)(x - (float)h);
      Bhi[base + e] = h;
      Blo[base + e] = l;
    }
  }
}

__device__ __forceinline__ float halfmax(float v) {
#pragma unroll
  for (int m = 1; m <= 8; m <<= 1) v = fmaxf(v, __shfl_xor(v, m, 32));
  return v;
}
__device__ __forceinline__ float halfsum(float v) {
#pragma unroll
  for (int m = 1; m <= 8; m <<= 1) v += __shfl_xor(v, m, 32);
  return v;
}

// logits = x_m @ x_c^T via split-bf16 WMMA (hi*hi + hi*lo + lo*hi), write logits to
// d_out and accumulate online per-row (max, expsum) partials per column chunk.
__global__ __launch_bounds__(256) void k_big(const bf16_t* __restrict__ Ahi,
                                             const bf16_t* __restrict__ Alo,
                                             const bf16_t* __restrict__ Bhi,
                                             const bf16_t* __restrict__ Blo,
                                             float* __restrict__ out,
                                             float* __restrict__ smax,
                                             float* __restrict__ ssum) {
  int chunk = threadIdx.x >> 5;   // 8 waves = 8 column chunks
  int lane  = threadIdx.x & 31;
  int rt    = blockIdx.x;         // row tile 0..1249
  int half = lane >> 4, q16 = lane & 15;

  v16bf ah[4], al[4];
#pragma unroll
  for (int qc = 0; qc < 4; ++qc) {
    size_t base = ((size_t)(rt * 4 + qc) * 32 + lane) * 16;
    ah[qc] = *(const v16bf*)(Ahi + base);
    al[qc] = *(const v16bf*)(Alo + base);
  }

  float rmax[8], rsum[8];
#pragma unroll
  for (int r = 0; r < 8; ++r) { rmax[r] = NEG_HUGE; rsum[r] = 0.f; }

  int t0 = chunk * TPC;
  int t1 = t0 + TPC; if (t1 > NT_COL) t1 = NT_COL;

  for (int nt = t0; nt < t1; ++nt) {
    v8f acc = {};
#pragma unroll
    for (int qc = 0; qc < 4; ++qc) {
      size_t bb = ((size_t)(nt * 4 + qc) * 32 + lane) * 16;
      v16bf bh = *(const v16bf*)(Bhi + bb);
      v16bf bl = *(const v16bf*)(Blo + bb);
      acc = __builtin_amdgcn_wmma_f32_16x16x32_bf16(false, ah[qc], false, bh,
                                                    (short)0, acc, false, false);
      acc = __builtin_amdgcn_wmma_f32_16x16x32_bf16(false, ah[qc], false, bl,
                                                    (short)0, acc, false, false);
      acc = __builtin_amdgcn_wmma_f32_16x16x32_bf16(false, al[qc], false, bh,
                                                    (short)0, acc, false, false);
    }
    int col = nt * 16 + q16;
    bool valid = col < C_N;
#pragma unroll
    for (int r = 0; r < 8; ++r) {
      if (valid) out[(size_t)(rt * 16 + half * 8 + r) * C_N + col] = acc[r];
    }
#pragma unroll
    for (int r = 0; r < 8; ++r) {
      float v  = valid ? acc[r] : NEG_HUGE;
      float tm = halfmax(v);                               // tile max (16-lane half)
      float p  = valid ? __expf(acc[r] - tm) : 0.f;
      float ts = halfsum(p);                               // tile expsum
      float nm = fmaxf(rmax[r], tm);
      rsum[r]  = rsum[r] * __expf(rmax[r] - nm) + ts * __expf(tm - nm);
      rmax[r]  = nm;
    }
  }
  if (q16 == 0) {
#pragma unroll
    for (int r = 0; r < 8; ++r) {
      int rowi = rt * 16 + half * 8 + r;
      smax[(size_t)rowi * NCHUNK + chunk] = rmax[r];
      ssum[(size_t)rowi * NCHUNK + chunk] = rsum[r];
    }
  }
}

// combine per-chunk partials -> per-row max and 1/denominator
__global__ void k_stats(const float* __restrict__ smax, const float* __restrict__ ssum,
                        float* __restrict__ rowM, float* __restrict__ rowI) {
  int row = blockIdx.x * blockDim.x + threadIdx.x;
  if (row >= M_N) return;
  float mx = NEG_HUGE;
#pragma unroll
  for (int c = 0; c < NCHUNK; ++c) mx = fmaxf(mx, smax[(size_t)row * NCHUNK + c]);
  float s = 0.f;
#pragma unroll
  for (int c = 0; c < NCHUNK; ++c)
    s += ssum[(size_t)row * NCHUNK + c] * __expf(smax[(size_t)row * NCHUNK + c] - mx);
  rowM[row] = mx;
  rowI[row] = 1.0f / s;
}

// out = exp(logit - rowmax) * rowinv   (float4 vectorized: 5000 % 4 == 0)
__global__ void k_norm(float* __restrict__ out, const float* __restrict__ rowM,
                       const float* __restrict__ rowI) {
  unsigned i = blockIdx.x * blockDim.x + threadIdx.x;
  if (i >= (unsigned)M_N * (C_N / 4)) return;
  int row = i / (C_N / 4);
  float4 v = ((float4*)out)[i];
  float mx = rowM[row], inv = rowI[row];
  v.x = __expf(v.x - mx) * inv;
  v.y = __expf(v.y - mx) * inv;
  v.z = __expf(v.z - mx) * inv;
  v.w = __expf(v.w - mx) * inv;
  ((float4*)out)[i] = v;
}

// ---------------- launch ----------------
extern "C" void kernel_launch(void* const* d_in, const int* in_sizes, int n_in,
                              void* d_out, int out_size, void* d_ws, size_t ws_size,
                              hipStream_t stream) {
  const float* m_feat  = (const float*)d_in[0];
  const float* c_feat  = (const float*)d_in[1];
  const float* own_val = (const float*)d_in[2];
  const float* W1 = (const float*)d_in[3];
  const float* b1 = (const float*)d_in[6];
  const float* W2 = (const float*)d_in[7];
  const float* b2 = (const float*)d_in[10];
  const float* W3 = (const float*)d_in[11];
  const float* b3 = (const float*)d_in[14];
  const int* own_row = (const int*)d_in[15];
  const int* own_col = (const int*)d_in[16];
  const int ne = in_sizes[2];

  float* out = (float*)d_out;
  char* ws = (char*)d_ws;
  float*  xm   = (float*)(ws + OFF_XM);
  float*  xc   = (float*)(ws + OFF_XC);
  bf16_t* Ahi  = (bf16_t*)(ws + OFF_AHI);
  bf16_t* Alo  = (bf16_t*)(ws + OFF_ALO);
  bf16_t* Bhi  = (bf16_t*)(ws + OFF_BHI);
  bf16_t* Blo  = (bf16_t*)(ws + OFF_BLO);
  float*  smax = (float*)(ws + OFF_SMAX);
  float*  ssum = (float*)(ws + OFF_SSUM);
  float*  rowM = (float*)(ws + OFF_ROWM);
  float*  rowI = (float*)(ws + OFF_ROWI);

  // 1) x_m = 0 ; scatter own edges ; += m_features@W1 + b1
  k_zero<<<(M_N * D_K + 255) / 256, 256, 0, stream>>>(xm, M_N * D_K);
  k_scatter_own<<<(ne + 7) / 8, 256, 0, stream>>>(c_feat, own_val, own_row, own_col, xm, ne);
  k_lin_xm<<<(NT_ROW * 8) / 4, 128, 0, stream>>>(m_feat, W1, b1, xm);

  // 2) x_c = c_features@(W2+W3) + b2 + b3   (2504 waves exactly)
  k_lin_xc<<<(NT_COL * 8) / 4, 128, 0, stream>>>(c_feat, W2, W3, b2, b3, xc);

  // 3) pack split-bf16 operands in WMMA register layout
  k_pack_A<<<(NT_ROW * 32 + 255) / 256, 256, 0, stream>>>(xm, Ahi, Alo);
  k_pack_B<<<(NT_COL * 32 + 255) / 256, 256, 0, stream>>>(xc, Bhi, Blo);

  // 4) big GEMM + online softmax partials
  k_big<<<NT_ROW, 256, 0, stream>>>(Ahi, Alo, Bhi, Blo, out, smax, ssum);

  // 5) combine stats, normalize
  k_stats<<<(M_N + 255) / 256, 256, 0, stream>>>(smax, ssum, rowM, rowI);
  k_norm<<<(M_N * (C_N / 4) + 255) / 256, 256, 0, stream>>>(out, rowM, rowI);
}
